// SlotAttnMulti_56255481643208
// MI455X (gfx1250) — matrix operand order repaired and checked by reading, not compiled, not run on hardware
//
#include <hip/hip_runtime.h>

// ---------------------------------------------------------------------------
// SlotAttention multi-view, CDNA5 (gfx1250, wave32) implementation.
// Heavy GEMMs use v_wmma_f32_16x16x32_bf16. Softmax kept in LDS (320KB/WGP).
// ---------------------------------------------------------------------------

typedef __attribute__((ext_vector_type(16))) __bf16 v16bf;
typedef __attribute__((ext_vector_type(8)))  float  v8f;

#define WMMA_BF16(A, B, C) \
  __builtin_amdgcn_wmma_f32_16x16x32_bf16(false, (A), false, (B), (short)0, (C), false, false)

// ---- constants --------------------------------------------------------------
#define NB   8
#define NV   4
#define NTOK 4096
#define NF   128
#define NS   7
#define QRY_ 64
#define DV_  64
#define DA_  128
#define DF_  192
#define NBV  32            // B*V
#define NROWS (NBV * NTOK) // 131072
#define COEF_ 0.125f
#define EPS_  1e-5f

// ---- wave reductions (wave32) ----------------------------------------------
__device__ inline float wsum32(float v) {
  for (int m = 16; m >= 1; m >>= 1) v += __shfl_xor(v, m, 32);
  return v;
}
__device__ inline float wmax32(float v) {
  for (int m = 16; m >= 1; m >>= 1) v = fmaxf(v, __shfl_xor(v, m, 32));
  return v;
}
__device__ inline float wsum16(float v) {  // reduce within 16-lane half
  for (int m = 8; m >= 1; m >>= 1) v += __shfl_xor(v, m, 32);
  return v;
}

// ---- WMMA fragment loaders (wave32 layouts per CDNA5 ISA 7.12.2) -----------
// A (16x32 bf16), row-major src[m*ld + k]:
//   lanes 0-15: M=lane, VGPR0-3 K=0..7, VGPR4-7 K=16..23
//   lanes16-31: M=lane-16, VGPR0-3 K=8..15, VGPR4-7 K=24..31
__device__ inline v16bf load_a_frag(const __bf16* src, int ld, int k0, int lane) {
  int m = lane & 15;
  int half = lane >> 4;
  v16bf a;
#pragma unroll
  for (int v = 0; v < 8; ++v) {
    int k = k0 + ((v >> 2) << 4) + half * 8 + ((v & 3) << 1);
    a[2 * v]     = src[(size_t)m * ld + k];
    a[2 * v + 1] = src[(size_t)m * ld + k + 1];
  }
  return a;
}

// B (32x16 bf16): element B[k][n] = src[k*sK + n*sN]
//   lanes 0-15: N=lane, VGPR0-3 K=0..7, VGPR4-7 K=16..23; lanes16-31: K=8..15, 24..31
__device__ inline v16bf load_b_frag(const __bf16* src, int sK, int sN,
                                    int k0, int n0, int lane) {
  int n = n0 + (lane & 15);
  int kb = k0 + ((lane >> 4) << 3);
  v16bf b;
#pragma unroll
  for (int v = 0; v < 8; ++v) {
    int k = kb + ((v >> 2) << 4) + ((v & 3) << 1);
    b[2 * v]     = src[(size_t)k * sK + (size_t)n * sN];
    b[2 * v + 1] = src[(size_t)(k + 1) * sK + (size_t)n * sN];
  }
  return b;
}

// C/D (16x16 f32): VGPR r -> M = r + (lane>>4)*8, N = lane&15
__device__ inline void store_c_rows(float* dst, int ld, int n0, int lane,
                                    v8f c, int maxrow) {
  int n = n0 + (lane & 15);
  int mb = (lane >> 4) * 8;
#pragma unroll
  for (int r = 0; r < 8; ++r) {
    int m = mb + r;
    if (m < maxrow) dst[(size_t)m * ld + n] = c[r];
  }
}

// ---------------------------------------------------------------------------
// Kernel 0: convert W_kv to bf16, initialize slot state from noise.
// ---------------------------------------------------------------------------
__global__ __launch_bounds__(256) void k_init(
    const float* W_kv, const float* noise_view, const float* noise_attr,
    const float* view_loc, const float* view_log_scl,
    const float* attr_loc, const float* attr_log_scl,
    __bf16* wkv_bf, float* slot_view, float* slot_attr) {
  int i = blockIdx.x * 256 + threadIdx.x;
  if (i < 256 * 128) wkv_bf[i] = (__bf16)W_kv[i];
  if (i < NB * NV * DV_) {
    int d = i & (DV_ - 1);
    slot_view[i] = view_loc[d] + __expf(view_log_scl[d]) * noise_view[i];
  }
  if (i < NB * NS * DA_) {
    int d = i & (DA_ - 1);
    slot_attr[i] = attr_loc[d] + __expf(attr_log_scl[d]) * noise_attr[i];
  }
}

// ---------------------------------------------------------------------------
// Kernel 1: kv = LN(x) @ W_kv^T ; split into key (bf16, *COEF) and val (bf16).
// One workgroup = 16 rows. 8 waves; LN via half-wave shuffles; WMMA K=128.
// Waves 0-1 always produce key columns, waves 2-7 val columns -> scalar branch.
// ---------------------------------------------------------------------------
__global__ __launch_bounds__(256) void k_kvproj(
    const float* __restrict__ x, const float* __restrict__ ln_w,
    const float* __restrict__ ln_b, const __bf16* __restrict__ wkv,
    __bf16* __restrict__ key_bf, __bf16* __restrict__ val_bf) {
  const size_t row0 = (size_t)blockIdx.x * 16;
  const int tid = threadIdx.x, w = tid >> 5, lane = tid & 31;
  __shared__ __bf16 a_s[16 * 128];

  { // --- LayerNorm: wave w handles rows 2w, 2w+1 (one per 16-lane half) ---
    int half = lane >> 4;
    int row = 2 * w + half;
    int c0 = (lane & 15) * 8;
    const float* xr = x + (row0 + row) * NF + c0;
    float v[8];
#pragma unroll
    for (int j = 0; j < 8; ++j) v[j] = xr[j];
    float s1 = 0.f, s2 = 0.f;
#pragma unroll
    for (int j = 0; j < 8; ++j) { s1 += v[j]; s2 += v[j] * v[j]; }
    s1 = wsum16(s1); s2 = wsum16(s2);
    float mean = s1 * (1.f / NF);
    float var = s2 * (1.f / NF) - mean * mean;
    float rstd = rsqrtf(var + EPS_);
#pragma unroll
    for (int j = 0; j < 8; ++j)
      a_s[row * 128 + c0 + j] =
          (__bf16)((v[j] - mean) * rstd * ln_w[c0 + j] + ln_b[c0 + j]);
  }
  __syncthreads();

  // --- 16x256 output tile; wave w owns 2 column tiles of 16 ---
  v16bf a0 = load_a_frag(a_s, 128, 0, lane);
  v16bf a1 = load_a_frag(a_s, 128, 32, lane);
  v16bf a2 = load_a_frag(a_s, 128, 64, lane);
  v16bf a3 = load_a_frag(a_s, 128, 96, lane);

#pragma unroll
  for (int tt = 0; tt < 2; ++tt) {
    int n0 = w * 32 + tt * 16;
    v8f c = {};
    // B[k][n] = W_kv[n][k] -> wkv[n*128 + k]
    c = WMMA_BF16(a0, load_b_frag(wkv, 1, 128, 0, n0, lane), c);
    c = WMMA_BF16(a1, load_b_frag(wkv, 1, 128, 32, n0, lane), c);
    c = WMMA_BF16(a2, load_b_frag(wkv, 1, 128, 64, n0, lane), c);
    c = WMMA_BF16(a3, load_b_frag(wkv, 1, 128, 96, n0, lane), c);

    // n0 is wave-uniform: force a scalar branch (no per-lane exec churn).
    int n0u = __builtin_amdgcn_readfirstlane(n0);
    int mb = (lane >> 4) * 8;
    if (n0u < QRY_) {
      int j = n0u + (lane & 15);
      __bf16* dst = key_bf + (row0 + mb) * QRY_ + j;
#pragma unroll
      for (int r = 0; r < 8; ++r) dst[(size_t)r * QRY_] = (__bf16)(c[r] * COEF_);
    } else {
      int j = (n0u - QRY_) + (lane & 15);
      __bf16* dst = val_bf + (row0 + mb) * DF_ + j;
#pragma unroll
      for (int r = 0; r < 8; ++r) dst[(size_t)r * DF_] = (__bf16)c[r];
    }
  }
}

// ---------------------------------------------------------------------------
// Kernel 2: build slot features, LN, q = LN(sf) @ W_q^T (bf16, padded to 16
// rows for WMMA A). One workgroup per (b,v).
// ---------------------------------------------------------------------------
__global__ __launch_bounds__(256) void k_qproj(
    const float* __restrict__ slot_view, const float* __restrict__ slot_attr,
    const float* __restrict__ ln_q_w, const float* __restrict__ ln_q_b,
    const float* __restrict__ W_q, float* __restrict__ sf_ws,
    __bf16* __restrict__ qb) {
  const int bv = blockIdx.x, b = bv >> 2;
  const int tid = threadIdx.x, w = tid >> 5, lane = tid & 31;
  __shared__ float sf_s[NS * DF_];
  __shared__ float ln_s[NS * DF_];

  for (int i = tid; i < NS * DF_; i += 256) {
    int s = i / DF_, d = i % DF_;
    float v = (d < DV_) ? slot_view[(size_t)bv * DV_ + d]
                        : slot_attr[((size_t)(b * NS + s)) * DA_ + (d - DV_)];
    sf_s[i] = v;
    sf_ws[(size_t)bv * NS * DF_ + i] = v;
  }
  __syncthreads();

  if (w < NS) {  // one wave per slot row
    float s1 = 0.f, s2 = 0.f;
    for (int d = lane; d < DF_; d += 32) {
      float v = sf_s[w * DF_ + d];
      s1 += v; s2 += v * v;
    }
    s1 = wsum32(s1); s2 = wsum32(s2);
    float mean = s1 * (1.f / DF_);
    float var = s2 * (1.f / DF_) - mean * mean;
    float rstd = rsqrtf(var + EPS_);
    for (int d = lane; d < DF_; d += 32)
      ln_s[w * DF_ + d] =
          (sf_s[w * DF_ + d] - mean) * rstd * ln_q_w[d] + ln_q_b[d];
  }
  __syncthreads();

  for (int i = tid; i < 16 * QRY_; i += 256) {
    int s = i / QRY_, j = i % QRY_;
    float acc = 0.f;
    if (s < NS) {
      const float* wr = &W_q[(size_t)j * DF_];
      for (int k = 0; k < DF_; ++k) acc += ln_s[s * DF_ + k] * wr[k];
    }
    qb[(size_t)bv * 16 * QRY_ + i] = (__bf16)acc;  // rows 7..15 zero
  }
}

// ---------------------------------------------------------------------------
// Kernel 3: attention for one (b,v) batch. logits (7x4096) live in LDS.
//   logits = q @ key^T   (WMMA, K=64)
//   logp   = logits - logsumexp_over_slots          (column pass)
//   attn   = softmax over tokens                    (row stats)
//   upd    = attn @ val  (WMMA, K=4096; A-fragments exp'd on the fly)
// ---------------------------------------------------------------------------
__global__ __launch_bounds__(256) void k_attn(
    const __bf16* __restrict__ qb, const __bf16* __restrict__ key_bf,
    const __bf16* __restrict__ val_bf, float* __restrict__ upd_ws) {
  const int bv = blockIdx.x;
  const int tid = threadIdx.x, w = tid >> 5, lane = tid & 31;
  __shared__ float logits_s[NS * NTOK];   // 114688 B (of 320KB LDS)
  __shared__ float rmax_s[8], rinv_s[8];

  const __bf16* q = qb + (size_t)bv * 16 * QRY_;
  const __bf16* key = key_bf + (size_t)bv * NTOK * QRY_;
  const __bf16* val = val_bf + (size_t)bv * NTOK * DF_;

  // ---- logits: A = q (16x64), B[k][n] = key[n*64+k] ----
  v16bf aq0 = load_a_frag(q, QRY_, 0, lane);
  v16bf aq1 = load_a_frag(q, QRY_, 32, lane);
  for (int t = w; t < NTOK / 16; t += 8) {
    int n0 = t * 16;
    v8f c = {};
    c = WMMA_BF16(aq0, load_b_frag(key, 1, QRY_, 0, n0, lane), c);
    c = WMMA_BF16(aq1, load_b_frag(key, 1, QRY_, 32, n0, lane), c);
    store_c_rows(logits_s, NTOK, n0, lane, c, NS);
  }
  __syncthreads();

  // ---- column pass: log_softmax over the 7 slots ----
  for (int n = tid; n < NTOK; n += 256) {
    float l[NS];
    float mx = -1e30f;
#pragma unroll
    for (int s = 0; s < NS; ++s) { l[s] = logits_s[s * NTOK + n]; mx = fmaxf(mx, l[s]); }
    float sum = 0.f;
#pragma unroll
    for (int s = 0; s < NS; ++s) sum += __expf(l[s] - mx);
    float lse = mx + __logf(sum);
#pragma unroll
    for (int s = 0; s < NS; ++s) logits_s[s * NTOK + n] = l[s] - lse;
  }
  __syncthreads();

  // ---- row stats: max and sum(exp) over 4096 tokens, one wave per slot ----
  if (w < NS) {
    float mx = -1e30f;
    for (int n = lane; n < NTOK; n += 32) mx = fmaxf(mx, logits_s[w * NTOK + n]);
    mx = wmax32(mx);
    float sum = 0.f;
    for (int n = lane; n < NTOK; n += 32) sum += __expf(logits_s[w * NTOK + n] - mx);
    sum = wsum32(sum);
    if (lane == 0) { rmax_s[w] = mx; rinv_s[w] = 1.f / sum; }
  }
  __syncthreads();

  // ---- upd = attn @ val : K=4096, 12 column tiles of DF=192 ----
  for (int t = w; t < DF_ / 16; t += 8) {
    int n0 = t * 16;
    v8f c = {};
#pragma unroll 2
    for (int kk = 0; kk < NTOK / 32; ++kk) {
      int k0 = kk * 32;
      // Prefetch the val rows two k-blocks ahead (streams from L2).
      {
        int kpf = (k0 + 64) & (NTOK - 1);
        __builtin_prefetch(val + (size_t)kpf * DF_ + n0 + (lane & 15), 0, 0);
      }
      // A fragment built on the fly: attn[m][k] = exp(logp - rmax) * rinv
      v16bf a;
      {
        int m = lane & 15;
        int half = lane >> 4;
        if (m < NS) {
          float mx = rmax_s[m], iv = rinv_s[m];
#pragma unroll
          for (int v = 0; v < 8; ++v) {
            int k = k0 + ((v >> 2) << 4) + half * 8 + ((v & 3) << 1);
            a[2 * v]     = (__bf16)(__expf(logits_s[m * NTOK + k] - mx) * iv);
            a[2 * v + 1] = (__bf16)(__expf(logits_s[m * NTOK + k + 1] - mx) * iv);
          }
        } else {
#pragma unroll
          for (int v = 0; v < 16; ++v) a[v] = (__bf16)0.f;
        }
      }
      v16bf bb = load_b_frag(val, DF_, 1, k0, n0, lane);  // B[k][n]=val[k*192+n]
      c = WMMA_BF16(a, bb, c);
    }
    store_c_rows(upd_ws + (size_t)bv * NS * DF_, DF_, n0, lane, c, NS);
  }
}

// ---------------------------------------------------------------------------
// Kernel 4: GRU cell + LN + residual MLP + mean pooling. One workgroup per b.
// ---------------------------------------------------------------------------
__global__ __launch_bounds__(256) void k_update(
    const float* __restrict__ sf_ws, const float* __restrict__ upd_ws,
    const float* __restrict__ W_ih, const float* __restrict__ W_hh,
    const float* __restrict__ b_ih, const float* __restrict__ b_hh,
    const float* __restrict__ ln_res_w, const float* __restrict__ ln_res_b,
    const float* __restrict__ W_res1, const float* __restrict__ b_res1,
    const float* __restrict__ W_res2, const float* __restrict__ b_res2,
    float* __restrict__ slot_view, float* __restrict__ slot_attr) {
  const int b = blockIdx.x, tid = threadIdx.x;
  const int w = tid >> 5, lane = tid & 31;
  const int R = NV * NS;  // 28 rows
  __shared__ float h_s[28 * DF_];    // sf (GRU hidden)
  __shared__ float x_s[28 * DF_];    // upd, later reused for sf4
  __shared__ float hn_s[28 * DF_];   // GRU output
  __shared__ float ln_s[28 * DF_];   // LN(h_new)
  __shared__ float hid_s[28 * 256];  // MLP hidden

  for (int i = tid; i < R * DF_; i += 256) {
    int row = i / DF_, d = i % DF_;
    int v = row / NS, s = row % NS;
    size_t src = ((size_t)(b * NV + v) * NS + s) * DF_ + d;
    h_s[i] = sf_ws[src];
    x_s[i] = upd_ws[src];
  }
  __syncthreads();

  // GRU: r,z,n gates per (row, j)
  for (int i = tid; i < R * DF_; i += 256) {
    int row = i / DF_, j = i % DF_;
    const float* xr = &x_s[row * DF_];
    const float* hr = &h_s[row * DF_];
    const float* wi0 = &W_ih[(size_t)j * DF_];
    const float* wi1 = &W_ih[(size_t)(DF_ + j) * DF_];
    const float* wi2 = &W_ih[(size_t)(2 * DF_ + j) * DF_];
    const float* wh0 = &W_hh[(size_t)j * DF_];
    const float* wh1 = &W_hh[(size_t)(DF_ + j) * DF_];
    const float* wh2 = &W_hh[(size_t)(2 * DF_ + j) * DF_];
    float gir = b_ih[j], giz = b_ih[DF_ + j], gin = b_ih[2 * DF_ + j];
    float ghr = b_hh[j], ghz = b_hh[DF_ + j], ghn = b_hh[2 * DF_ + j];
    for (int k = 0; k < DF_; ++k) {
      float xv = xr[k], hv = hr[k];
      gir += xv * wi0[k]; giz += xv * wi1[k]; gin += xv * wi2[k];
      ghr += hv * wh0[k]; ghz += hv * wh1[k]; ghn += hv * wh2[k];
    }
    float r = 1.f / (1.f + __expf(-(gir + ghr)));
    float z = 1.f / (1.f + __expf(-(giz + ghz)));
    float n = tanhf(gin + r * ghn);
    hn_s[i] = (1.f - z) * n + z * hr[j];
  }
  __syncthreads();

  // LN over each of 28 rows
  for (int row = w; row < R; row += 8) {
    float s1 = 0.f, s2 = 0.f;
    for (int d = lane; d < DF_; d += 32) {
      float v = hn_s[row * DF_ + d];
      s1 += v; s2 += v * v;
    }
    s1 = wsum32(s1); s2 = wsum32(s2);
    float mean = s1 * (1.f / DF_);
    float var = s2 * (1.f / DF_) - mean * mean;
    float rstd = rsqrtf(var + EPS_);
    for (int d = lane; d < DF_; d += 32)
      ln_s[row * DF_ + d] =
          (hn_s[row * DF_ + d] - mean) * rstd * ln_res_w[d] + ln_res_b[d];
  }
  __syncthreads();

  // MLP hidden: relu(ln @ W_res1^T + b)
  for (int i = tid; i < R * 256; i += 256) {
    int row = i / 256, jj = i % 256;
    const float* wr = &W_res1[(size_t)jj * DF_];
    float acc = b_res1[jj];
    for (int k = 0; k < DF_; ++k) acc += ln_s[row * DF_ + k] * wr[k];
    hid_s[i] = fmaxf(acc, 0.f);
  }
  __syncthreads();

  // MLP out + residual -> sf4 (reuse x_s)
  for (int i = tid; i < R * DF_; i += 256) {
    int row = i / DF_, j = i % DF_;
    const float* wr = &W_res2[(size_t)j * 256];
    float acc = b_res2[j];
    for (int k = 0; k < 256; ++k) acc += hid_s[row * 256 + k] * wr[k];
    x_s[i] = hn_s[i] + acc;
  }
  __syncthreads();

  // slot_view[b,v,:] = mean over s of sf4[..., :64]
  for (int i = tid; i < NV * DV_; i += 256) {
    int v = i / DV_, d = i % DV_;
    float acc = 0.f;
    for (int s = 0; s < NS; ++s) acc += x_s[(v * NS + s) * DF_ + d];
    slot_view[((size_t)(b * NV + v)) * DV_ + d] = acc * (1.f / NS);
  }
  // slot_attr[b,s,:] = mean over v of sf4[..., 64:]
  for (int i = tid; i < NS * DA_; i += 256) {
    int s = i / DA_, d = i % DA_;
    float acc = 0.f;
    for (int v = 0; v < NV; ++v) acc += x_s[(v * NS + s) * DF_ + DV_ + d];
    slot_attr[((size_t)(b * NS + s)) * DA_ + d] = acc * (1.f / NV);
  }
}

// ---------------------------------------------------------------------------
// Host-side launch
// ---------------------------------------------------------------------------
extern "C" void kernel_launch(void* const* d_in, const int* in_sizes, int n_in,
                              void* d_out, int out_size, void* d_ws,
                              size_t ws_size, hipStream_t stream) {
  (void)in_sizes; (void)n_in; (void)out_size; (void)ws_size;
  const float* x            = (const float*)d_in[0];
  const float* noise_view   = (const float*)d_in[2];
  const float* noise_attr   = (const float*)d_in[3];
  const float* view_loc     = (const float*)d_in[4];
  const float* view_log_scl = (const float*)d_in[5];
  const float* attr_loc     = (const float*)d_in[6];
  const float* attr_log_scl = (const float*)d_in[7];
  const float* ln_kv_w      = (const float*)d_in[8];
  const float* ln_kv_b      = (const float*)d_in[9];
  const float* W_kv         = (const float*)d_in[10];
  const float* ln_q_w       = (const float*)d_in[11];
  const float* ln_q_b       = (const float*)d_in[12];
  const float* W_q          = (const float*)d_in[13];
  const float* W_ih         = (const float*)d_in[14];
  const float* W_hh         = (const float*)d_in[15];
  const float* b_ih         = (const float*)d_in[16];
  const float* b_hh         = (const float*)d_in[17];
  const float* ln_res_w     = (const float*)d_in[18];
  const float* ln_res_b     = (const float*)d_in[19];
  const float* W_res1       = (const float*)d_in[20];
  const float* b_res1       = (const float*)d_in[21];
  const float* W_res2       = (const float*)d_in[22];
  const float* b_res2       = (const float*)d_in[23];

  // workspace layout (bytes)
  char* ws = (char*)d_ws;
  float*  slot_view = (float*)(ws + 0);          //  8*4*64  f32 (output head)
  float*  slot_attr = (float*)(ws + 8192);       //  8*7*128 f32 (output head)
  __bf16* wkv_bf    = (__bf16*)(ws + 36864);     //  256*128 bf16
  __bf16* qb        = (__bf16*)(ws + 126976);    //  32*16*64 bf16 (padded q)
  float*  sf_ws     = (float*)(ws + 192512);     //  32*7*192 f32
  float*  upd_ws    = (float*)(ws + 364544);     //  32*7*192 f32
  __bf16* key_bf    = (__bf16*)(ws + 536576);    //  32*4096*64 bf16
  __bf16* val_bf    = (__bf16*)(ws + 17313792);  //  32*4096*192 bf16

  k_init<<<128, 256, 0, stream>>>(W_kv, noise_view, noise_attr, view_loc,
                                  view_log_scl, attr_loc, attr_log_scl,
                                  wkv_bf, slot_view, slot_attr);

  k_kvproj<<<NROWS / 16, 256, 0, stream>>>(x, ln_kv_w, ln_kv_b, wkv_bf,
                                           key_bf, val_bf);

  for (int step = 0; step < 3; ++step) {
    k_qproj<<<NBV, 256, 0, stream>>>(slot_view, slot_attr, ln_q_w, ln_q_b,
                                     W_q, sf_ws, qb);
    k_attn<<<NBV, 256, 0, stream>>>(qb, key_bf, val_bf, upd_ws);
    k_update<<<NB, 256, 0, stream>>>(sf_ws, upd_ws, W_ih, W_hh, b_ih, b_hh,
                                     ln_res_w, ln_res_b, W_res1, b_res1,
                                     W_res2, b_res2, slot_view, slot_attr);
  }

  // outputs are contiguous at the head of the workspace: (slot_view, slot_attr)
  hipMemcpyAsync(d_out, ws, (size_t)(2048 + 7168) * sizeof(float),
                 hipMemcpyDeviceToDevice, stream);
}
